// MaskedCausalAttention_5205500362932
// MI455X (gfx1250) — compile-verified, matrix operand order checked by reading, not run on hardware
//
#include <hip/hip_runtime.h>
#include <hip/hip_bf16.h>
#include <math.h>

// ---------------------------------------------------------------------------
// MaskedCausalAttention for MI455X (gfx1250, wave32)
// bf16 WMMA everywhere, TDM double-buffered LDS staging, ds_swizzle softmax.
// B=4, T=1024, C=1024, NH=16, Dh=64
// ---------------------------------------------------------------------------

typedef __bf16 bf16;
typedef __attribute__((ext_vector_type(16))) __bf16 v16bf;
typedef __attribute__((ext_vector_type(8)))  __bf16 v8bf;
typedef __attribute__((ext_vector_type(4)))  __bf16 v4bf;
typedef __attribute__((ext_vector_type(8)))  float  v8f;
typedef __attribute__((ext_vector_type(4)))  float  v4f;
typedef __attribute__((ext_vector_type(4)))  unsigned int u32x4;
typedef __attribute__((ext_vector_type(8)))  unsigned int u32x8;

#define WMMA_BF16(a, b, c) \
  __builtin_amdgcn_wmma_f32_16x16x32_bf16(false, (a), false, (b), (short)0, (c), false, false)

static __device__ __forceinline__ v16bf cat8(v8bf lo, v8bf hi) {
  return __builtin_shufflevector(lo, hi, 0,1,2,3,4,5,6,7,8,9,10,11,12,13,14,15);
}

// xor-shuffle within 32 lanes via immediate-pattern ds_swizzle (no index math)
template <int XM>
static __device__ __forceinline__ float swz_xor(float v) {
  return __int_as_float(
      __builtin_amdgcn_ds_swizzle(__float_as_int(v), (XM << 10) | 0x1f));
}

static constexpr int Bb = 4, Ts = 1024, Cc = 1024, NH = 16, Dh = 64;

// ---------------------------------------------------------------------------
// TDM 2-D tile copy: global [tile_h x tile_w] bf16 (row stride ldRow elems)
// -> packed row-major LDS. D# per CDNA5 ISA ch.8. Tracked by TENSORcnt.
// ---------------------------------------------------------------------------
static __device__ __forceinline__ void tdm_load_tile_bf16(
    unsigned int lds_byte_addr, const bf16* gptr,
    unsigned int tile_w, unsigned int tile_h, unsigned int ldRow,
    unsigned int tensor_h) {
  const unsigned long long ga = (unsigned long long)(uintptr_t)gptr;
  u32x4 g0;
  g0[0] = 1u;                                            // count=1, user mode
  g0[1] = lds_byte_addr;                                 // lds_addr
  g0[2] = (unsigned int)ga;                              // global_addr[31:0]
  g0[3] = (unsigned int)((ga >> 32) & 0x01FFFFFFu)       // global_addr[56:32]
          | (2u << 30);                                  // type=2 (image)
  u32x8 g1;
  g1[0] = (1u << 16);                                    // data_size=1 -> 2 bytes
  g1[1] = (tile_w << 16);                                // tensor_dim0[15:0]
  g1[2] = (tensor_h << 16);                              // dim0 hi | tensor_dim1 lo
  g1[3] = (tile_w << 16);                                // dim1 hi | tile_dim0
  g1[4] = tile_h;                                        // tile_dim1 | tile_dim2=0
  g1[5] = ldRow;                                         // tensor_dim0_stride lo32
  g1[6] = 0u;
  g1[7] = 0u;
  asm volatile("tensor_load_to_lds %0, %1" :: "s"(g0), "s"(g1) : "memory");
}

// ---------------------------------------------------------------------------
// f32 -> bf16 bulk convert (one-time; frees the GEMM hot loop of VALU cvt)
// ---------------------------------------------------------------------------
__global__ __launch_bounds__(256)
void cvt_f32_to_bf16(const float* __restrict__ src, bf16* __restrict__ dst, int n4) {
  const int i = blockIdx.x * 256 + threadIdx.x;
  if (i < n4) {
    const v4f v = ((const v4f*)src)[i];
    v4bf o;
#pragma unroll
    for (int e = 0; e < 4; ++e) o[e] = (bf16)v[e];
    ((v4bf*)dst)[i] = o;
  }
}

// ---------------------------------------------------------------------------
// GEMM: Y[M,N] = (X[M,K] @ W[N,K]^T + bias[N]) * oscale   (bf16 in, WMMA)
// Block = 256 threads (8 waves, 4x2). Tile 128x128, K-step 32.
// TDM double-buffers the 128x32 A/B tiles into LDS; per K-step each wave does
// 12 ds_load_b128 fragment reads + 8 v_wmma_f32_16x16x32_bf16.
// MODE: 0 = f32 out [M,N], 1 = bf16 out [M,N], 2 = bf16 out transposed [B][C][T]
// ---------------------------------------------------------------------------
template <int MODE>
__global__ __launch_bounds__(256)
void gemm_xWT(const bf16* __restrict__ X, const bf16* __restrict__ W,
              const float* __restrict__ bias, void* __restrict__ Yv,
              int M, int N, int K, float oscale) {
  __shared__ bf16 As[2][128][32];   // [buf][m][k]
  __shared__ bf16 Bs[2][128][32];   // [buf][n][k]

  const int tid  = threadIdx.x;
  const int wave = tid >> 5;
  const int lane = tid & 31;
  const int half = lane >> 4;
  const int l15  = lane & 15;
  const int kofs = half * 8;
  const int wm   = wave >> 1;       // 0..3 (M)
  const int wn   = wave & 1;        // 0..1 (N)
  const int bm   = blockIdx.y * 128;
  const int bn   = blockIdx.x * 128;

  const unsigned int ldsA0 = (unsigned int)(uintptr_t)(void*)&As[0][0][0];
  const unsigned int ldsA1 = (unsigned int)(uintptr_t)(void*)&As[1][0][0];
  const unsigned int ldsB0 = (unsigned int)(uintptr_t)(void*)&Bs[0][0][0];
  const unsigned int ldsB1 = (unsigned int)(uintptr_t)(void*)&Bs[1][0][0];

  v8f acc[2][4] = {};
  const int nT = K / 32;

  if (wave == 0) {
    tdm_load_tile_bf16(ldsA0, X + (size_t)bm * K, 32, 128, K, M);
    tdm_load_tile_bf16(ldsB0, W + (size_t)bn * K, 32, 128, K, N);
  }

  for (int it = 0; it < nT; ++it) {
    const int buf = it & 1;
    if (wave == 0) {
      if (it + 1 < nT) {   // prefetch next tile into the other buffer
        const int k1 = (it + 1) * 32;
        tdm_load_tile_bf16(buf ? ldsA0 : ldsA1, X + (size_t)bm * K + k1, 32, 128, K, M);
        tdm_load_tile_bf16(buf ? ldsB0 : ldsB1, W + (size_t)bn * K + k1, 32, 128, K, N);
        __builtin_amdgcn_s_wait_tensorcnt(2);   // current tile's 2 ops done
      } else {
        __builtin_amdgcn_s_wait_tensorcnt(0);
      }
    }
    __syncthreads();   // current buffer visible to all waves

    v16bf a[2];
#pragma unroll
    for (int mt = 0; mt < 2; ++mt) {
      const int arow = wm * 32 + mt * 16 + l15;
      a[mt] = cat8(*(const v8bf*)&As[buf][arow][kofs],
                   *(const v8bf*)&As[buf][arow][16 + kofs]);
    }
#pragma unroll
    for (int ct = 0; ct < 4; ++ct) {
      const int bcol = wn * 64 + ct * 16 + l15;
      const v16bf b = cat8(*(const v8bf*)&Bs[buf][bcol][kofs],
                           *(const v8bf*)&Bs[buf][bcol][16 + kofs]);
      acc[0][ct] = WMMA_BF16(a[0], b, acc[0][ct]);
      acc[1][ct] = WMMA_BF16(a[1], b, acc[1][ct]);
    }
    __syncthreads();   // everyone done with this buffer before it is refilled
  }

  // Store. C layout: VGPR i -> M = i + 8*half ; N = l15.
#pragma unroll
  for (int mt = 0; mt < 2; ++mt) {
#pragma unroll
    for (int ct = 0; ct < 4; ++ct) {
      const int n  = bn + wn * 64 + ct * 16 + l15;
      const float bb = bias[n];
#pragma unroll
      for (int i = 0; i < 8; ++i) {
        const int m = bm + wm * 32 + mt * 16 + i + half * 8;
        const float v = (acc[mt][ct][i] + bb) * oscale;
        if constexpr (MODE == 0) {
          ((float*)Yv)[(size_t)m * N + n] = v;
        } else if constexpr (MODE == 1) {
          ((bf16*)Yv)[(size_t)m * N + n] = (bf16)v;
        } else {  // transposed bf16: [B][C][T], b = m/Ts, t = m%Ts
          const int btok = m >> 10, t = m & (Ts - 1);
          ((bf16*)Yv)[((size_t)btok * Cc + n) * Ts + t] = (bf16)v;
        }
      }
    }
  }
}

// ---------------------------------------------------------------------------
// Fused causal attention, flash style. Block = 4 waves = 64 query rows of one
// (batch, head). Q pre-scaled by 1/sqrt(D) at projection time. K tiles
// [32 keys x 64 d] and V^T tiles [64 d x 32 keys] are TDM double-buffered.
// Per 32-key chunk: 4 score WMMAs, ds_swizzle online softmax, 4 PV WMMAs.
// ---------------------------------------------------------------------------
__global__ __launch_bounds__(128)
void attn_fused(const bf16* __restrict__ Q, const bf16* __restrict__ Kg,
                const bf16* __restrict__ Vt, const unsigned char* __restrict__ amask,
                bf16* __restrict__ Out) {
  __shared__ bf16 Ks[2][32][64];    // [buf][key][d]
  __shared__ bf16 Vs[2][64][32];    // [buf][d][key]   (V transposed)
  __shared__ bf16 Ps[4][16][32];    // per-wave P tile [q][key]

  const int tid  = threadIdx.x;
  const int wave = tid >> 5;
  const int lane = tid & 31;
  const int half = lane >> 4;
  const int l15  = lane & 15;
  const int kofs = half * 8;
  const float NEG_INF = -__builtin_inff();

  const int blocksPerSeq = Ts / 64;
  const int qblk = blockIdx.x % blocksPerSeq;
  const int bh   = blockIdx.x / blocksPerSeq;
  const int h    = bh % NH;
  const int b    = bh / NH;

  const bf16* Qb  = Q  + (size_t)b * Ts * Cc + h * Dh;             // [T][C]
  const bf16* Kb  = Kg + (size_t)b * Ts * Cc + h * Dh;             // [T][C]
  const bf16* Vtb = Vt + ((size_t)b * Cc + h * Dh) * Ts;           // [C][T]
  const unsigned char* mb = amask + (size_t)b * Ts;

  const unsigned int ldsK0 = (unsigned int)(uintptr_t)(void*)&Ks[0][0][0];
  const unsigned int ldsK1 = (unsigned int)(uintptr_t)(void*)&Ks[1][0][0];
  const unsigned int ldsV0 = (unsigned int)(uintptr_t)(void*)&Vs[0][0][0];
  const unsigned int ldsV1 = (unsigned int)(uintptr_t)(void*)&Vs[1][0][0];

  const int q0 = qblk * 64 + wave * 16;

  // Preload Q fragments (d 0..31, 32..63).
  const bf16* qrow = Qb + (size_t)(q0 + l15) * Cc;
  const v16bf aQlo = cat8(*(const v8bf*)(qrow + kofs),      *(const v8bf*)(qrow + 16 + kofs));
  const v16bf aQhi = cat8(*(const v8bf*)(qrow + 32 + kofs), *(const v8bf*)(qrow + 48 + kofs));

  v8f  Oacc[4] = {};
  float mrow[8], lrow[8];
#pragma unroll
  for (int i = 0; i < 8; ++i) { mrow[i] = NEG_INF; lrow[i] = 0.f; }

  const int nSteps = (qblk * 64 + 64) / 32;   // 32-key chunks up to causal bound
  if (wave == 0) {
    tdm_load_tile_bf16(ldsK0, Kb,  Dh, 32, Cc, Ts);
    tdm_load_tile_bf16(ldsV0, Vtb, 32, Dh, Ts, Dh);
  }

  for (int st = 0; st < nSteps; ++st) {
    const int buf = st & 1;
    const int j0  = st * 32;
    if (wave == 0) {
      if (st + 1 < nSteps) {
        const int j1 = j0 + 32;
        tdm_load_tile_bf16(buf ? ldsK0 : ldsK1, Kb + (size_t)j1 * Cc, Dh, 32, Cc, Ts);
        tdm_load_tile_bf16(buf ? ldsV0 : ldsV1, Vtb + j1,             32, Dh, Ts, Dh);
        __builtin_amdgcn_s_wait_tensorcnt(2);
      } else {
        __builtin_amdgcn_s_wait_tensorcnt(0);
      }
    }
    __syncthreads();

    if (j0 <= q0) {   // wave-uniform causal activity test (j0 > q0 => all masked)
      // ---- scores (Q already carries 1/sqrt(D)) ----
      v8f S[2];
#pragma unroll
      for (int jt = 0; jt < 2; ++jt) {
        const bf16* krow = &Ks[buf][jt * 16 + l15][0];
        const v16bf bKlo = cat8(*(const v8bf*)(krow + kofs),      *(const v8bf*)(krow + 16 + kofs));
        const v16bf bKhi = cat8(*(const v8bf*)(krow + 32 + kofs), *(const v8bf*)(krow + 48 + kofs));
        v8f s = {};
        s = WMMA_BF16(aQlo, bKlo, s);
        s = WMMA_BF16(aQhi, bKhi, s);
        const int col  = j0 + jt * 16 + l15;
        const bool mok = (mb[col] != 0);
        if (j0 + 31 > q0) {        // diagonal chunk: causal + padding per element
#pragma unroll
          for (int i = 0; i < 8; ++i) {
            const int row = q0 + i + half * 8;
            S[jt][i] = (mok && col <= row) ? s[i] : NEG_INF;
          }
        } else {                   // interior chunk: padding mask only
#pragma unroll
          for (int i = 0; i < 8; ++i) S[jt][i] = mok ? s[i] : NEG_INF;
        }
      }

      // ---- online softmax: rowmax via ds_swizzle xor tree ----
      float msafe[8], scl[8];
#pragma unroll
      for (int i = 0; i < 8; ++i) {
        float t = fmaxf(S[0][i], S[1][i]);
        t = fmaxf(t, swz_xor<1>(t));
        t = fmaxf(t, swz_xor<2>(t));
        t = fmaxf(t, swz_xor<4>(t));
        t = fmaxf(t, swz_xor<8>(t));
        const float mold = mrow[i];
        const float mn   = fmaxf(mold, t);
        msafe[i] = (mn == NEG_INF) ? 0.f : mn;
        scl[i]   = __expf(mold - msafe[i]);
        mrow[i]  = mn;
      }

      // ---- p = exp(s-m); spill P bf16 to LDS for A-fragment re-layout ----
#pragma unroll
      for (int jt = 0; jt < 2; ++jt) {
#pragma unroll
        for (int i = 0; i < 8; ++i) {
          const float p = __expf(S[jt][i] - msafe[i]);
          S[jt][i] = p;
          Ps[wave][i + half * 8][jt * 16 + l15] = (bf16)p;
        }
      }

      // ---- row sums + rescale running state ----
#pragma unroll
      for (int i = 0; i < 8; ++i) {
        float rs = S[0][i] + S[1][i];
        rs += swz_xor<1>(rs);
        rs += swz_xor<2>(rs);
        rs += swz_xor<4>(rs);
        rs += swz_xor<8>(rs);
        lrow[i] = lrow[i] * scl[i] + rs;
#pragma unroll
        for (int dt = 0; dt < 4; ++dt) Oacc[dt][i] *= scl[i];
      }

      asm volatile("s_wait_dscnt 0" ::: "memory");  // P stores -> loads (same wave)

      // ---- O += P V : both fragments now contiguous ds_load_b128 ----
      const bf16* prow = &Ps[wave][l15][0];
      const v16bf aP = cat8(*(const v8bf*)(prow + kofs), *(const v8bf*)(prow + 16 + kofs));
#pragma unroll
      for (int dt = 0; dt < 4; ++dt) {
        const bf16* vrow = &Vs[buf][dt * 16 + l15][0];   // B[k=key][n=d], keys contiguous
        const v16bf bV = cat8(*(const v8bf*)(vrow + kofs), *(const v8bf*)(vrow + 16 + kofs));
        Oacc[dt] = WMMA_BF16(aP, bV, Oacc[dt]);
      }
    }
    __syncthreads();   // all waves done with buffer before refill
  }

  // ---- normalize, write bf16 attn output [B,T,C] ----
  bf16* Ob = Out + (size_t)b * Ts * Cc + h * Dh;
#pragma unroll
  for (int i = 0; i < 8; ++i) {
    const float rl = (lrow[i] > 0.f) ? (1.f / lrow[i]) : 0.f;
    const int row  = q0 + i + half * 8;
#pragma unroll
    for (int dt = 0; dt < 4; ++dt)
      Ob[(size_t)row * Cc + dt * 16 + l15] = (bf16)(Oacc[dt][i] * rl);
  }
}

// ---------------------------------------------------------------------------
// Launch. Workspace (48 MB):
//   xbf 8M | Qbf 8M | Kbf 8M | Vtbf 8M | attnbf 8M | Wq/Wk/Wv/Wo bf 2M each
// ---------------------------------------------------------------------------
extern "C" void kernel_launch(void* const* d_in, const int* in_sizes, int n_in,
                              void* d_out, int out_size, void* d_ws, size_t ws_size,
                              hipStream_t stream) {
  const float*         x     = (const float*)d_in[0];
  const unsigned char* amask = (const unsigned char*)d_in[1];
  const float* Wq = (const float*)d_in[2];
  const float* bq = (const float*)d_in[3];
  const float* Wk = (const float*)d_in[4];
  const float* bk = (const float*)d_in[5];
  const float* Wv = (const float*)d_in[6];
  const float* bv = (const float*)d_in[7];
  const float* Wo = (const float*)d_in[8];
  const float* bo = (const float*)d_in[9];
  float* out = (float*)d_out;

  char* ws = (char*)d_ws;
  const size_t MB = 1024 * 1024;
  bf16* xbf    = (bf16*)(ws);
  bf16* Qbf    = (bf16*)(ws + 8  * MB);
  bf16* Kbf    = (bf16*)(ws + 16 * MB);
  bf16* Vtbf   = (bf16*)(ws + 24 * MB);
  bf16* attnbf = (bf16*)(ws + 32 * MB);
  bf16* Wqbf   = (bf16*)(ws + 40 * MB);
  bf16* Wkbf   = (bf16*)(ws + 42 * MB);
  bf16* Wvbf   = (bf16*)(ws + 44 * MB);
  bf16* Wobf   = (bf16*)(ws + 46 * MB);

  const int M = Bb * Ts;                       // 4096
  const int nx4 = (M * Cc) / 4, nw4 = (Cc * Cc) / 4;
  cvt_f32_to_bf16<<<(nx4 + 255) / 256, 256, 0, stream>>>(x,  xbf,  nx4);
  cvt_f32_to_bf16<<<(nw4 + 255) / 256, 256, 0, stream>>>(Wq, Wqbf, nw4);
  cvt_f32_to_bf16<<<(nw4 + 255) / 256, 256, 0, stream>>>(Wk, Wkbf, nw4);
  cvt_f32_to_bf16<<<(nw4 + 255) / 256, 256, 0, stream>>>(Wv, Wvbf, nw4);
  cvt_f32_to_bf16<<<(nw4 + 255) / 256, 256, 0, stream>>>(Wo, Wobf, nw4);

  dim3 gg(Cc / 128, M / 128);                  // 8 x 32 blocks, 256 threads
  const float qscale = 0.125f;                 // 1/sqrt(Dh), folded into Q
  gemm_xWT<1><<<gg, 256, 0, stream>>>(xbf, Wqbf, bq, Qbf,  M, Cc, Cc, qscale);
  gemm_xWT<1><<<gg, 256, 0, stream>>>(xbf, Wkbf, bk, Kbf,  M, Cc, Cc, 1.0f);
  gemm_xWT<2><<<gg, 256, 0, stream>>>(xbf, Wvbf, bv, Vtbf, M, Cc, Cc, 1.0f);

  attn_fused<<<Bb * NH * (Ts / 64), 128, 0, stream>>>(Qbf, Kbf, Vtbf, amask, attnbf);

  gemm_xWT<0><<<gg, 256, 0, stream>>>(attnbf, Wobf, bo, out, M, Cc, Cc, 1.0f);
}
